// lstm_seq2seq_82927228551686
// MI455X (gfx1250) — matrix operand order repaired and checked by reading, not compiled, not run on hardware
//
#include <hip/hip_runtime.h>
#include <hip/hip_bf16.h>

#define B_   256
#define S_   512
#define I_   64
#define H_   128
#define T_   256
#define LDH  136      // padded halves per LDS h row (128 + 8) -> 272B stride, 16B aligned

typedef __attribute__((ext_vector_type(16))) _Float16 v16h;
typedef __attribute__((ext_vector_type(8)))  _Float16 v8h;
typedef __attribute__((ext_vector_type(8)))  float    v8f;

// ---- hardware transcendentals (CDNA5 TRANS32 tanh) ----
static __device__ __forceinline__ float tanh_hw(float x) {
#if __has_builtin(__builtin_amdgcn_tanhf)
  return __builtin_amdgcn_tanhf(x);
#else
  return tanhf(x);
#endif
}
static __device__ __forceinline__ float sigmoid_hw(float x) {
  return __builtin_fmaf(0.5f, tanh_hw(0.5f * x), 0.5f);
}

static __device__ __forceinline__ v8f wmma16(v16h a, v16h b, v8f c) {
  return __builtin_amdgcn_wmma_f32_16x16x32_f16(
      false, a, false, b, (short)0, c, false, false);
}

// ---- A fragment (16x32 f16): lane(0-15)=row, hi=lane/16;
// halves [k0..k0+7] then [k0+16..k0+23], k0 = kc*32 + hi*8
static __device__ __forceinline__ v16h a_frag_lds(const _Float16* row, int kc, int hi) {
  const v8h* p0 = (const v8h*)(row + kc * 32 + hi * 8);
  const v8h* p1 = (const v8h*)(row + kc * 32 + hi * 8 + 16);
  v8h lo = *p0, hiV = *p1;
  v16h r;
#pragma unroll
  for (int j = 0; j < 8; ++j) { r[j] = lo[j]; r[8 + j] = hiV[j]; }
  return r;
}

static __device__ __forceinline__ v16h a_frag_f32(const float* row, int kc, int hi) {
  const float4* p0 = (const float4*)(row + kc * 32 + hi * 8);
  const float4* p1 = (const float4*)(row + kc * 32 + hi * 8 + 16);
  float4 f0 = p0[0], f1 = p0[1], f2 = p1[0], f3 = p1[1];
  v16h r;
  r[0]=(_Float16)f0.x;  r[1]=(_Float16)f0.y;  r[2]=(_Float16)f0.z;  r[3]=(_Float16)f0.w;
  r[4]=(_Float16)f1.x;  r[5]=(_Float16)f1.y;  r[6]=(_Float16)f1.z;  r[7]=(_Float16)f1.w;
  r[8]=(_Float16)f2.x;  r[9]=(_Float16)f2.y;  r[10]=(_Float16)f2.z; r[11]=(_Float16)f2.w;
  r[12]=(_Float16)f3.x; r[13]=(_Float16)f3.y; r[14]=(_Float16)f3.z; r[15]=(_Float16)f3.w;
  return r;
}

// ---- B fragment (32x16 f16): lane(0-15)=col n; K [k0..k0+7] then [k0+16..k0+23], k0 = kc*32 + hi*8
static __device__ __forceinline__ v16h b_frag_f32(const float* row, int kc, int hi) {
  const float4* p = (const float4*)(row + kc * 32 + hi * 8);
  v16h r;
#pragma unroll
  for (int q = 0; q < 4; ++q) {
    float4 f = p[q + (q >> 1) * 2];
    r[4*q+0]=(_Float16)f.x; r[4*q+1]=(_Float16)f.y;
    r[4*q+2]=(_Float16)f.z; r[4*q+3]=(_Float16)f.w;
  }
  return r;
}

// ---- xz fragment (C-layout, 8 f32 per lane, contiguous) ----
struct ZF { float4 a, b; };

static __device__ __forceinline__ ZF load_z(const float* xz, int t, int mb,
                                            int tile, int lane) {
  size_t off = (((size_t)t * 512) + (size_t)mb * 32 + tile) * 256 + lane * 8;
  const float4* p = (const float4*)(xz + off);
  ZF z; z.a = p[0]; z.b = p[1]; return z;
}

static __device__ __forceinline__ void z_to_acc(const ZF& z, v8f& a) {
  a[0]=z.a.x; a[1]=z.a.y; a[2]=z.a.z; a[3]=z.a.w;
  a[4]=z.b.x; a[5]=z.b.y; a[6]=z.b.z; a[7]=z.b.w;
}

// one LSTM matmul step: acc[g] = z[g] + h_prev @ W_hh^T (gate tiles)
static __device__ __forceinline__ void mm_step(const _Float16* rrow,
                                               const v16h Bw[4][4],
                                               const ZF zc[4], v8f acc[4], int hi) {
  v16h A0 = a_frag_lds(rrow, 0, hi);
  v16h A1 = a_frag_lds(rrow, 1, hi);
  v16h A2 = a_frag_lds(rrow, 2, hi);
  v16h A3 = a_frag_lds(rrow, 3, hi);
#pragma unroll
  for (int g = 0; g < 4; ++g) {
    z_to_acc(zc[g], acc[g]);
    acc[g] = wmma16(A0, Bw[g][0], acc[g]);
    acc[g] = wmma16(A1, Bw[g][1], acc[g]);
    acc[g] = wmma16(A2, Bw[g][2], acc[g]);
    acc[g] = wmma16(A3, Bw[g][3], acc[g]);
  }
}

// gate nonlinearity + c update + h write (into write buffer)
static __device__ __forceinline__ void gate_update(const v8f acc[4], v8f& c,
                                                   _Float16* wbuf, int hi, int hc) {
#pragma unroll
  for (int j = 0; j < 8; ++j) {
    float iv = sigmoid_hw(acc[0][j]);
    float fv = sigmoid_hw(acc[1][j]);
    float gv = tanh_hw(acc[2][j]);
    float ov = sigmoid_hw(acc[3][j]);
    float cv = __builtin_fmaf(fv, c[j], iv * gv);
    c[j] = cv;
    float hv = ov * tanh_hw(cv);
    wbuf[(j + 8 * hi) * LDH + hc] = (_Float16)hv;   // C/D layout: m = v + 8*(lane/16)
  }
}

// ============================================================================
// Kernel 1: xz[t] = x[:, t_src, :] @ W_ih^T + (b_ih + b_hh), WMMA C-fragment order
// ============================================================================
__global__ __launch_bounds__(256) void xz_gemm_kernel(
    const float* __restrict__ x, const float* __restrict__ W_ih,
    const float* __restrict__ b_ih, const float* __restrict__ b_hh,
    float* __restrict__ xz, int dec_mode) {
  int wg   = blockIdx.x;
  int t    = wg >> 4;
  int mb   = wg & 15;
  int lane = threadIdx.x & 31;
  int wv   = threadIdx.x >> 5;
  int hi   = lane >> 4;
  int ln16 = lane & 15;

  int t_src = dec_mode ? ((t == 0) ? (S_ - 1) : (S_ - t)) : t;

  const float* arow = x + ((size_t)(mb * 16 + ln16) * S_ + t_src) * I_;
  v16h A0 = a_frag_f32(arow, 0, hi);
  v16h A1 = a_frag_f32(arow, 1, hi);

#pragma unroll
  for (int g = 0; g < 4; ++g) {
    int nt = wv + 8 * g;
    int n  = nt * 16 + ln16;
    const float* brow = W_ih + (size_t)n * I_;
    v16h Bf0 = b_frag_f32(brow, 0, hi);
    v16h Bf1 = b_frag_f32(brow, 1, hi);
    float bias = b_ih[n] + b_hh[n];
    v8f acc;
#pragma unroll
    for (int j = 0; j < 8; ++j) acc[j] = bias;
    acc = wmma16(A0, Bf0, acc);
    acc = wmma16(A1, Bf1, acc);
    size_t off = (((size_t)t * 512) + (size_t)mb * 32 + nt) * 256 + lane * 8;
    float4* p = (float4*)(xz + off);
    p[0] = make_float4(acc[0], acc[1], acc[2], acc[3]);
    p[1] = make_float4(acc[4], acc[5], acc[6], acc[7]);
  }
}

// ============================================================================
// Kernel 2: persistent recurrence, double-buffered h in LDS (1 barrier/step),
// z fragments software-pipelined one step ahead, W_hh resident in VGPRs.
// ============================================================================
__global__ __launch_bounds__(256) void lstm_rec_kernel(
    const float* __restrict__ enc_W_hh, const float* __restrict__ dec_W_hh,
    const float* __restrict__ lin_W,    const float* __restrict__ lin_b,
    const float* __restrict__ xz_enc,   const float* __restrict__ xz_dec,
    float* __restrict__ out) {
  __shared__ _Float16 hlds[2][16 * LDH];

  int mb   = blockIdx.x;
  int lane = threadIdx.x & 31;
  int wv   = threadIdx.x >> 5;
  int hi   = lane >> 4;
  int ln16 = lane & 15;
  int hc   = wv * 16 + ln16;

  // resident encoder W_hh fragments
  v16h Bw[4][4];
#pragma unroll
  for (int g = 0; g < 4; ++g) {
    const float* brow = enc_W_hh + (size_t)((wv + 8 * g) * 16 + ln16) * H_;
#pragma unroll
    for (int kc = 0; kc < 4; ++kc) Bw[g][kc] = b_frag_f32(brow, kc, hi);
  }

  for (int i = threadIdx.x; i < 2 * 16 * LDH; i += 256)
    (&hlds[0][0])[i] = (_Float16)0.f;
  v8f c;
#pragma unroll
  for (int j = 0; j < 8; ++j) c[j] = 0.f;

  const _Float16* row0 = &hlds[0][0] + (size_t)ln16 * LDH;
  const _Float16* row1 = &hlds[1][0] + (size_t)ln16 * LDH;
  _Float16* buf0 = &hlds[0][0];
  _Float16* buf1 = &hlds[1][0];

  ZF zc[4], zn[4];
#pragma unroll
  for (int g = 0; g < 4; ++g) zc[g] = load_z(xz_enc, 0, mb, wv + 8 * g, lane);
  __syncthreads();

  // -------------------- encoder: 512 dependent steps --------------------
  for (int t = 0; t < S_; t += 2) {
    // even step: read buf0, write buf1
#pragma unroll
    for (int g = 0; g < 4; ++g) zn[g] = load_z(xz_enc, t + 1, mb, wv + 8 * g, lane);
    {
      v8f acc[4];
      mm_step(row0, Bw, zc, acc, hi);
      gate_update(acc, c, buf1, hi, hc);
    }
#pragma unroll
    for (int g = 0; g < 4; ++g) zc[g] = zn[g];
    __syncthreads();

    // odd step: read buf1, write buf0
    int tn = (t + 2 < S_) ? (t + 2) : (S_ - 1);   // harmless tail refetch
#pragma unroll
    for (int g = 0; g < 4; ++g) zn[g] = load_z(xz_enc, tn, mb, wv + 8 * g, lane);
    {
      v8f acc[4];
      mm_step(row1, Bw, zc, acc, hi);
      gate_update(acc, c, buf0, hi, hc);
    }
#pragma unroll
    for (int g = 0; g < 4; ++g) zc[g] = zn[g];
    __syncthreads();
  }
  // h_enc now in buf0; c carried in registers.

  // switch resident fragments: decoder W_hh + linear head
#pragma unroll
  for (int g = 0; g < 4; ++g) {
    const float* brow = dec_W_hh + (size_t)((wv + 8 * g) * 16 + ln16) * H_;
#pragma unroll
    for (int kc = 0; kc < 4; ++kc) Bw[g][kc] = b_frag_f32(brow, kc, hi);
  }
  v16h Lb[4];
  float lbias = 0.f;
  if (wv < 4) {
    const float* brow = lin_W + (size_t)(wv * 16 + ln16) * H_;
#pragma unroll
    for (int kc = 0; kc < 4; ++kc) Lb[kc] = b_frag_f32(brow, kc, hi);
    lbias = lin_b[wv * 16 + ln16];
  }

#pragma unroll
  for (int g = 0; g < 4; ++g) zc[g] = load_z(xz_dec, 0, mb, wv + 8 * g, lane);

  // -------------------- decoder: 256 dependent steps --------------------
  for (int t = 0; t < T_; t += 2) {
#pragma unroll
    for (int half = 0; half < 2; ++half) {
      int tcur = t + half;
      const _Float16* rrow = half ? row1 : row0;
      _Float16* wbuf       = half ? buf0 : buf1;
      const _Float16* prow = half ? row0 : row1;   // row of the buffer just written

      int tn = (tcur + 1 < T_) ? (tcur + 1) : (T_ - 1);
#pragma unroll
      for (int g = 0; g < 4; ++g) zn[g] = load_z(xz_dec, tn, mb, wv + 8 * g, lane);
      {
        v8f acc[4];
        mm_step(rrow, Bw, zc, acc, hi);
        gate_update(acc, c, wbuf, hi, hc);
      }
#pragma unroll
      for (int g = 0; g < 4; ++g) zc[g] = zn[g];
      __syncthreads();   // h(tcur) complete in wbuf

      // linear head on h(tcur): waves 0-3, wave-uniform branch (EXEC all-1s inside)
      if (wv < 4) {
        v16h N0 = a_frag_lds(prow, 0, hi);
        v16h N1 = a_frag_lds(prow, 1, hi);
        v16h N2 = a_frag_lds(prow, 2, hi);
        v16h N3 = a_frag_lds(prow, 3, hi);
        v8f o;
#pragma unroll
        for (int j = 0; j < 8; ++j) o[j] = lbias;
        o = wmma16(N0, Lb[0], o);
        o = wmma16(N1, Lb[1], o);
        o = wmma16(N2, Lb[2], o);
        o = wmma16(N3, Lb[3], o);
        int nc = wv * 16 + ln16;
#pragma unroll
        for (int j = 0; j < 8; ++j) {
          int m = j + 8 * hi;
          out[((size_t)(mb * 16 + m) * T_ + tcur) * I_ + nc] = o[j];
        }
      }
    }
  }
}

// ============================================================================
extern "C" void kernel_launch(void* const* d_in, const int* in_sizes, int n_in,
                              void* d_out, int out_size, void* d_ws, size_t ws_size,
                              hipStream_t stream) {
  (void)in_sizes; (void)n_in; (void)out_size; (void)ws_size;
  const float* x        = (const float*)d_in[0];
  const float* enc_W_ih = (const float*)d_in[1];
  const float* enc_W_hh = (const float*)d_in[2];
  const float* enc_b_ih = (const float*)d_in[3];
  const float* enc_b_hh = (const float*)d_in[4];
  const float* dec_W_ih = (const float*)d_in[5];
  const float* dec_W_hh = (const float*)d_in[6];
  const float* dec_b_ih = (const float*)d_in[7];
  const float* dec_b_hh = (const float*)d_in[8];
  const float* lin_W    = (const float*)d_in[9];
  const float* lin_b    = (const float*)d_in[10];

  float* xz_enc = (float*)d_ws;                         // S*512*256 f32
  float* xz_dec = xz_enc + (size_t)S_ * 512 * 256;      // T*512*256 f32

  xz_gemm_kernel<<<S_ * 16, 256, 0, stream>>>(x, enc_W_ih, enc_b_ih, enc_b_hh, xz_enc, 0);
  xz_gemm_kernel<<<T_ * 16, 256, 0, stream>>>(x, dec_W_ih, dec_b_ih, dec_b_hh, xz_dec, 1);

  lstm_rec_kernel<<<16, 256, 0, stream>>>(enc_W_hh, dec_W_hh, lin_W, lin_b,
                                          xz_enc, xz_dec, (float*)d_out);
}
